// VQVAE_76373108457703
// MI455X (gfx1250) — compile-verified, matrix operand order checked by reading, not run on hardware
//
#include <hip/hip_runtime.h>
#include <hip/hip_bf16.h>
#include <math.h>

// ---------------------------------------------------------------------------
// Types / WMMA helpers (CDNA5 / gfx1250, wave32)
// ---------------------------------------------------------------------------
typedef _Float16 half_t;
typedef __attribute__((ext_vector_type(16))) _Float16 v16h;
typedef __attribute__((ext_vector_type(8)))  float    v8f;
typedef int v4i __attribute__((vector_size(16)));
typedef __attribute__((address_space(1))) v4i* gv4i_p;   // global int4*
typedef __attribute__((address_space(3))) v4i* lv4i_p;   // LDS int4*

#define WMMA_F16(a, b, c) \
  __builtin_amdgcn_wmma_f32_16x16x32_f16(false, (a), false, (b), (short)0, (c), false, false)

// Async Global->LDS staging (CDNA5): guarded so the compile never breaks.
#if defined(__has_builtin)
# if __has_builtin(__builtin_amdgcn_global_load_async_to_lds_b128) && \
     __has_builtin(__builtin_amdgcn_s_wait_asynccnt)
#  define HAVE_ASYNC_LDS 1
# endif
#endif
#ifndef HAVE_ASYNC_LDS
# define HAVE_ASYNC_LDS 0
#endif

// Fixed problem sizes
constexpr int  SEQL   = 3584;
constexpr int  BATCH  = 8;
constexpr int  MROWS  = SEQL * BATCH;   // 28672, row order r = s*8 + b
constexpr int  DM     = 512;
constexpr int  GATES  = 2048;
constexpr int  KCB    = 1024;           // codebook entries

// A fragment (16x32 f16): lane L<16 -> row M=L, K in [k0..k0+7] and [k0+16..k0+23];
// lane L>=16 -> row M=L-16, K base +8.
__device__ __forceinline__ v16h load_a_frag(const half_t* __restrict__ A, int lda,
                                            int row0, int kt, int lane) {
  const half_t* p = A + (size_t)(row0 + (lane & 15)) * lda + kt * 32 + ((lane & 16) ? 8 : 0);
  v16h a;
#pragma unroll
  for (int j = 0; j < 8; ++j) { a[j] = p[j]; a[j + 8] = p[j + 16]; }
  return a;
}

// A fragment out of a 16x32 row-major LDS tile
__device__ __forceinline__ v16h lds_a_frag(const half_t* buf, int lane) {
  const half_t* p = buf + (lane & 15) * 32 + ((lane & 16) ? 8 : 0);
  v16h a;
#pragma unroll
  for (int j = 0; j < 8; ++j) { a[j] = p[j]; a[j + 8] = p[j + 16]; }
  return a;
}

// Packed B: tile (nt, kt) is 512 halves; lane's 16 halves contiguous at lane*16.
__device__ __forceinline__ v16h load_b_frag(const half_t* __restrict__ Bp, int KT,
                                            int nt, int kt, int lane) {
  const half_t* p = Bp + ((size_t)nt * KT + kt) * 512 + lane * 16;
  return *(const v16h*)p;
}

// Stage one 16x32 f16 A tile (1KB) into LDS: threads 0..63 move 16B each.
__device__ __forceinline__ void stage_a_tile(const half_t* __restrict__ A, int lda,
                                             int row0, int kt, half_t* buf, int tid) {
  if (tid < 64) {
    int row = tid >> 2, chunk = tid & 3;
    const half_t* src = A + (size_t)(row0 + row) * lda + kt * 32 + chunk * 8;
    half_t* dst = buf + row * 32 + chunk * 8;
#if HAVE_ASYNC_LDS
    __builtin_amdgcn_global_load_async_to_lds_b128((gv4i_p)src, (lv4i_p)dst, 0, 0);
#else
    *reinterpret_cast<float4*>(dst) = *reinterpret_cast<const float4*>(src);
#endif
  }
}

__device__ __forceinline__ void async_stage_wait() {
#if HAVE_ASYNC_LDS
  __builtin_amdgcn_s_wait_asynccnt(0);
#endif
}

__device__ __forceinline__ float sigmf(float x) { return 1.0f / (1.0f + expf(-x)); }

// ---------------------------------------------------------------------------
// Weight pack: B[k][n] (opt. transposed source), zero-padded, into fragment layout
// ---------------------------------------------------------------------------
__global__ void pack_b_kernel(const float* __restrict__ W, half_t* __restrict__ out,
                              int KT, int NT, int srcK, int srcN, int ld, int transB) {
  int t = blockIdx.x * 256 + threadIdx.x;
  int total = KT * NT * 32;
  if (t >= total) return;
  int lane = t & 31, bi = t >> 5;
  int kt = bi % KT, nt = bi / KT;
  int n  = nt * 16 + (lane & 15);
  int kb = kt * 32 + ((lane & 16) ? 8 : 0);
  half_t* o = out + (size_t)bi * 512 + lane * 16;
#pragma unroll
  for (int j = 0; j < 8; ++j) {
    int k0 = kb + j, k1 = kb + 16 + j;
    float v0 = 0.f, v1 = 0.f;
    if (n < srcN) {
      if (k0 < srcK) v0 = transB ? W[(size_t)n * ld + k0] : W[(size_t)k0 * ld + n];
      if (k1 < srcK) v1 = transB ? W[(size_t)n * ld + k1] : W[(size_t)k1 * ld + n];
    }
    o[j] = (half_t)v0; o[j + 8] = (half_t)v1;
  }
}

// ---------------------------------------------------------------------------
// Embedding gather + concat -> X0 f16 [MROWS x 1248], row r = s*8 + b
// ---------------------------------------------------------------------------
__global__ void gather_kernel(const int* __restrict__ tokens,
                              const float* e0, const float* e1, const float* e2,
                              const float* e3, const float* e4, const float* e5,
                              const float* e6, half_t* __restrict__ X0) {
  size_t idx = (size_t)blockIdx.x * 256 + threadIdx.x;
  const size_t total = (size_t)MROWS * 1248;
  if (idx >= total) return;
  int r = (int)(idx / 1248), f = (int)(idx % 1248);
  int s = r >> 3, b = r & 7;
  const int offs[8] = {0, 128, 384, 448, 480, 992, 1120, 1248};
  const int esz[7]  = {128, 256, 64, 32, 512, 128, 128};
  const float* embs[7] = {e0, e1, e2, e3, e4, e5, e6};
  int i = 0;
  while (f >= offs[i + 1]) ++i;
  int tok = tokens[((size_t)b * SEQL + s) * 7 + i];
  X0[idx] = (half_t)embs[i][(size_t)tok * esz[i] + (f - offs[i])];
}

// ---------------------------------------------------------------------------
// Generic f16 WMMA GEMM: C[M,N] = A[M,K] * Bpacked + bias ; f32 and/or f16 out.
// grid = (M/16, ceil(N/64)), block = 128 (4 waves, one 16x16 N-tile each).
// The A tile is shared by all 4 waves: staged through a double-buffered LDS
// slab with async Global->LDS copies (ASYNCcnt), consumed via ds_load_b128.
// ---------------------------------------------------------------------------
__global__ void gemm_kernel(const half_t* __restrict__ A, const half_t* __restrict__ Bp,
                            const float* __restrict__ bias, float* __restrict__ C,
                            half_t* __restrict__ C16, int N, int KT) {
  __shared__ half_t aT[2][16 * 32];
  const int tid  = threadIdx.x;
  const int wave = tid >> 5, lane = tid & 31;
  const int row0 = blockIdx.x * 16;
  const int nt   = blockIdx.y * 4 + wave;
  const bool active = (nt * 16 < N);
  const int n = nt * 16 + (lane & 15);
  v8f acc = {};
  if (active && bias) {
    float bv = bias[n];
#pragma unroll
    for (int j = 0; j < 8; ++j) acc[j] = bv;
  }

  const int lda = KT * 32;
  stage_a_tile(A, lda, row0, 0, aT[0], tid);
  async_stage_wait();
  __syncthreads();

  for (int kt = 0; kt < KT; ++kt) {
    if (kt + 1 < KT) stage_a_tile(A, lda, row0, kt + 1, aT[(kt + 1) & 1], tid);
    if (active) {
      if (kt + 1 < KT)
        __builtin_prefetch(Bp + ((size_t)nt * KT + kt + 1) * 512 + lane * 16, 0, 1);
      v16h a = lds_a_frag(aT[kt & 1], lane);
      v16h b = load_b_frag(Bp, KT, nt, kt, lane);
      acc = WMMA_F16(a, b, acc);
    }
    async_stage_wait();
    __syncthreads();
  }

  if (active) {
#pragma unroll
    for (int j = 0; j < 8; ++j) {
      int m = (lane & 16) ? j + 8 : j;
      size_t off = (size_t)(row0 + m) * N + n;
      if (C)   C[off]   = acc[j];
      if (C16) C16[off] = (half_t)acc[j];
    }
  }
}

// ---------------------------------------------------------------------------
// Device-wide barrier (persistent LSTM kernel): sense-reversal on gen counter
// ---------------------------------------------------------------------------
__device__ __forceinline__ void grid_sync(unsigned* cnt, unsigned* gen, unsigned nwg) {
  __threadfence();
  __syncthreads();
  if (threadIdx.x == 0) {
    unsigned g = __hip_atomic_load(gen, __ATOMIC_RELAXED, __HIP_MEMORY_SCOPE_AGENT);
    unsigned prev = __hip_atomic_fetch_add(cnt, 1u, __ATOMIC_ACQ_REL, __HIP_MEMORY_SCOPE_AGENT);
    if (prev == nwg - 1u) {
      __hip_atomic_store(cnt, 0u, __ATOMIC_RELAXED, __HIP_MEMORY_SCOPE_AGENT);
      __hip_atomic_fetch_add(gen, 1u, __ATOMIC_RELEASE, __HIP_MEMORY_SCOPE_AGENT);
    } else {
      while (__hip_atomic_load(gen, __ATOMIC_ACQUIRE, __HIP_MEMORY_SCOPE_AGENT) == g) {
        __builtin_amdgcn_s_sleep(2);
      }
    }
  }
  __syncthreads();
}

// ---------------------------------------------------------------------------
// Persistent LSTM. 32 WGs x 128 threads. WG w owns h columns [16w,16w+16).
// Wave g computes gate-g preact tile via 16-WMMA K-chain over h_prev (f16).
// c-state lives in LDS for the whole sequence; h published as f16 for next step.
// ---------------------------------------------------------------------------
__global__ void lstm_kernel(const float* __restrict__ xW, const half_t* __restrict__ WhhP,
                            float* __restrict__ out, half_t* __restrict__ hbuf,
                            unsigned* cnt, unsigned* gen) {
  __shared__ float preact[4][16][16];
  __shared__ float cstate[16][16];
  const int tid = threadIdx.x;
  const int wave = tid >> 5, lane = tid & 31;
  const int w = blockIdx.x;                   // 0..31
  for (int e = tid; e < 256; e += 128) cstate[e >> 4][e & 15] = 0.f;
  __syncthreads();

  const int ncol  = wave * 512 + w * 16;      // gate-column base for this wave
  const int ntile = ncol >> 4;                // packed-B tile index (KT = 16)

  for (int s = 0; s < SEQL; ++s) {
    const int row0 = s * 8;
    v8f acc = {};
#pragma unroll
    for (int j = 0; j < 8; ++j) {
      int m = (lane & 16) ? j + 8 : j;
      acc[j] = (m < 8) ? xW[(size_t)(row0 + m) * GATES + ncol + (lane & 15)] : 0.f;
    }
#pragma unroll 4
    for (int kt = 0; kt < 16; ++kt) {
      v16h a = load_a_frag(hbuf, DM, 0, kt, lane);
      v16h b = load_b_frag(WhhP, 16, ntile, kt, lane);
      acc = WMMA_F16(a, b, acc);
    }
#pragma unroll
    for (int j = 0; j < 8; ++j) {
      int m = (lane & 16) ? j + 8 : j;
      preact[wave][m][lane & 15] = acc[j];
    }
    __syncthreads();
    for (int e = tid; e < 256; e += 128) {
      int m = e >> 4, c = e & 15;
      float iv = preact[0][m][c], fv = preact[1][m][c];
      float gv = preact[2][m][c], ov = preact[3][m][c];
      float cs = sigmf(fv) * cstate[m][c] + sigmf(iv) * tanhf(gv);
      cstate[m][c] = cs;
      float h = sigmf(ov) * tanhf(cs);
      hbuf[(size_t)m * DM + w * 16 + c] = (half_t)h;
      if (m < 8) out[(size_t)(row0 + m) * DM + w * 16 + c] = h;
    }
    grid_sync(cnt, gen, gridDim.x);
  }
}

// ---------------------------------------------------------------------------
// f32 -> f16 convert, optionally applying the every-30th-step encoder mask
// ---------------------------------------------------------------------------
__global__ void cvt_f16_kernel(const float* __restrict__ in, half_t* __restrict__ out,
                               int apply_mask) {
  size_t idx = (size_t)blockIdx.x * 256 + threadIdx.x;
  const size_t total = (size_t)MROWS * DM;
  if (idx >= total) return;
  float v = in[idx];
  if (apply_mask) {
    int s = (int)(idx >> 9) >> 3;
    if (s % 30) v = 0.f;
  }
  out[idx] = (half_t)v;
}

// codebook row norms
__global__ void cbnorm_kernel(const float* __restrict__ cb, float* __restrict__ nrm) {
  int n = blockIdx.x * 256 + threadIdx.x;
  if (n >= KCB) return;
  float s = 0.f;
  for (int k = 0; k < DM; ++k) { float v = cb[(size_t)n * DM + k]; s += v * v; }
  nrm[n] = s;
}

__global__ void vecadd_kernel(const float* a, const float* b, float* o, int n) {
  int i = blockIdx.x * 256 + threadIdx.x;
  if (i < n) o[i] = a[i] + b[i];
}

__global__ void head_bias_kernel(const float* pb, float* o, int v, int vp) {
  int i = blockIdx.x * 256 + threadIdx.x;
  if (i < vp) o[i] = (i < v) ? pb[i] : -1.0e30f;
}

// ---------------------------------------------------------------------------
// VQ argmin: score(n) = ||c_n||^2 - 2 e.c_n  (||e||^2 constant per row).
// A fragments hoisted into registers once (invariant across 16 N-tiles).
// ---------------------------------------------------------------------------
__global__ void vq_kernel(const half_t* __restrict__ encm, const half_t* __restrict__ cbP,
                          const float* __restrict__ cbnorm, int* __restrict__ inds) {
  __shared__ float s_s[4][16][16];
  __shared__ int   s_i[4][16][16];
  const int tid = threadIdx.x, wave = tid >> 5, lane = tid & 31;
  const int row0 = blockIdx.x * 16;
  v16h afr[16];
#pragma unroll
  for (int kt = 0; kt < 16; ++kt) afr[kt] = load_a_frag(encm, DM, row0, kt, lane);
  float best[8]; int bidx[8];
#pragma unroll
  for (int j = 0; j < 8; ++j) { best[j] = 3.4e38f; bidx[j] = 0; }
  for (int t = 0; t < 16; ++t) {
    int nt = wave * 16 + t;
    v8f acc = {};
#pragma unroll
    for (int kt = 0; kt < 16; ++kt) {
      v16h b = load_b_frag(cbP, 16, nt, kt, lane);
      acc = WMMA_F16(afr[kt], b, acc);
    }
    int n = nt * 16 + (lane & 15);
    float cn = cbnorm[n];
#pragma unroll
    for (int j = 0; j < 8; ++j) {
      float sc = cn - 2.0f * acc[j];
      if (sc < best[j] || (sc == best[j] && n < bidx[j])) { best[j] = sc; bidx[j] = n; }
    }
  }
#pragma unroll
  for (int j = 0; j < 8; ++j) {
    int m = (lane & 16) ? j + 8 : j;
    s_s[wave][m][lane & 15] = best[j];
    s_i[wave][m][lane & 15] = bidx[j];
  }
  __syncthreads();
  if (tid < 16) {
    float bs = 3.4e38f; int bi = 0;
    for (int wv = 0; wv < 4; ++wv)
      for (int c = 0; c < 16; ++c) {
        float sc = s_s[wv][tid][c]; int ii = s_i[wv][tid][c];
        if (sc < bs || (sc == bs && ii < bi)) { bs = sc; bi = ii; }
      }
    inds[row0 + tid] = bi;
  }
}

// quant gather (forward value of straight-through) + vq MSE accumulation
__global__ void quant_loss_kernel(const float* __restrict__ enc, const float* __restrict__ cb,
                                  const int* __restrict__ inds, half_t* __restrict__ q16,
                                  float* __restrict__ vqacc) {
  __shared__ float red[256];
  size_t idx = (size_t)blockIdx.x * 256 + threadIdx.x;
  const size_t total = (size_t)MROWS * DM;
  float d2 = 0.f;
  if (idx < total) {
    int r = (int)(idx >> 9), c = (int)(idx & 511);
    int s = r >> 3;
    float e = enc[idx] * ((s % 30) == 0 ? 1.f : 0.f);
    float q = cb[(size_t)inds[r] * DM + c];
    q16[idx] = (half_t)q;
    float d = q - e; d2 = d * d;
  }
  red[threadIdx.x] = d2;
  __syncthreads();
  for (int st = 128; st > 0; st >>= 1) {
    if ((int)threadIdx.x < st) red[threadIdx.x] += red[threadIdx.x + st];
    __syncthreads();
  }
  if (threadIdx.x == 0) atomicAdd(vqacc, red[0]);
}

// cross-entropy over one head's logits (padded cols carry -1e30 bias -> exp=0)
__global__ void ce_kernel(const float* __restrict__ logits, int vp,
                          const int* __restrict__ tokens, int head, float* __restrict__ acc) {
  const int wave = threadIdx.x >> 5, lane = threadIdx.x & 31;
  const int row = blockIdx.x * 8 + wave;
  const float* lp = logits + (size_t)row * vp;
  float mx = -3.4e38f;
  for (int c = lane; c < vp; c += 32) mx = fmaxf(mx, lp[c]);
  for (int off = 16; off > 0; off >>= 1) mx = fmaxf(mx, __shfl_xor(mx, off, 32));
  float se = 0.f;
  for (int c = lane; c < vp; c += 32) se += expf(lp[c] - mx);
  for (int off = 16; off > 0; off >>= 1) se += __shfl_xor(se, off, 32);
  if (lane == 0) {
    int s = row >> 3, b = row & 7;
    int label = tokens[((size_t)b * SEQL + s) * 7 + head];
    atomicAdd(acc, -(lp[label] - mx - logf(se)));
  }
}

// permute dec [s*8+b, c] -> d_out [b, s, c] ; append scalar losses
__global__ void output_kernel(const float* __restrict__ decbuf, const float* __restrict__ vqacc,
                              const float* __restrict__ ceacc, float* __restrict__ out) {
  size_t idx = (size_t)blockIdx.x * 256 + threadIdx.x;
  const size_t total = (size_t)MROWS * DM;
  if (idx < total) {
    int r = (int)(idx >> 9), c = (int)(idx & 511);
    int s = r >> 3, b = r & 7;
    out[((size_t)b * SEQL + s) * DM + c] = decbuf[idx];
  }
  if (blockIdx.x == 0 && threadIdx.x == 0) {
    out[total]     = ceacc[0] / (7.0f * (float)MROWS);
    out[total + 1] = 1.25f * vqacc[0] / (float)((size_t)MROWS * DM);
  }
}

// ---------------------------------------------------------------------------
// Host driver
// ---------------------------------------------------------------------------
extern "C" void kernel_launch(void* const* d_in, const int* in_sizes, int n_in,
                              void* d_out, int out_size, void* d_ws, size_t ws_size,
                              hipStream_t stream) {
  (void)in_sizes; (void)n_in; (void)out_size; (void)ws_size;
  const int*   tokens = (const int*)d_in[0];
  const float* embp[7];
  for (int i = 0; i < 7; ++i) embp[i] = (const float*)d_in[1 + i];
  const float* W_in  = (const float*)d_in[8];
  const float* b_in  = (const float*)d_in[9];
  const float* eWih  = (const float*)d_in[10];
  const float* eWhh  = (const float*)d_in[11];
  const float* ebih  = (const float*)d_in[12];
  const float* ebhh  = (const float*)d_in[13];
  const float* dWih  = (const float*)d_in[14];
  const float* dWhh  = (const float*)d_in[15];
  const float* dbih  = (const float*)d_in[16];
  const float* dbhh  = (const float*)d_in[17];
  const float* cbook = (const float*)d_in[18];

  const int vocab[7] = {56, 135, 18, 4, 87, 18, 25};
  const int vpad[7]  = {64, 144, 32, 16, 96, 32, 32};

  char* W = (char*)d_ws;
  size_t off = 0;
  auto A_ = [&](size_t bytes) { size_t o = off; off += (bytes + 255) & ~(size_t)255; return o; };
  size_t o_X0   = A_((size_t)MROWS * 1248 * 2);
  size_t o_WinP = A_((size_t)1248 * 512 * 2);
  size_t o_eWihP = A_((size_t)512 * 2048 * 2);
  size_t o_eWhhP = A_((size_t)512 * 2048 * 2);
  size_t o_dWihP = A_((size_t)512 * 2048 * 2);
  size_t o_dWhhP = A_((size_t)512 * 2048 * 2);
  size_t o_cbP  = A_((size_t)512 * 1024 * 2);
  size_t o_hP[7];
  for (int i = 0; i < 7; ++i) o_hP[i] = A_((size_t)512 * vpad[i] * 2);
  size_t o_encB = A_(2048 * 4);
  size_t o_decB = A_(2048 * 4);
  size_t o_hB[7];
  for (int i = 0; i < 7; ++i) o_hB[i] = A_((size_t)vpad[i] * 4);
  size_t o_cbn  = A_(1024 * 4);
  size_t o_x16  = A_((size_t)MROWS * DM * 2);
  size_t o_xW   = A_((size_t)MROWS * GATES * 4);
  size_t o_enc  = A_((size_t)MROWS * DM * 4);
  size_t o_em16 = A_((size_t)MROWS * DM * 2);
  size_t o_inds = A_((size_t)MROWS * 4);
  size_t o_q16  = A_((size_t)MROWS * DM * 2);
  size_t o_dec  = A_((size_t)MROWS * DM * 4);
  size_t o_d16  = A_((size_t)MROWS * DM * 2);
  size_t o_log  = A_((size_t)MROWS * 512 * 4);
  size_t o_sync = A_(256);
  size_t o_acc  = A_(256);
  size_t o_hbuf = A_((size_t)16 * DM * 2);

  unsigned* syncp = (unsigned*)(W + o_sync);
  float*    accp  = (float*)(W + o_acc);     // [0]=vq mse sum, [1]=ce sum
  half_t*   hbuf  = (half_t*)(W + o_hbuf);

  (void)hipMemsetAsync(W + o_sync, 0, 256, stream);
  (void)hipMemsetAsync(W + o_acc,  0, 256, stream);
  (void)hipMemsetAsync(W + o_hbuf, 0, (size_t)16 * DM * 2, stream);

  auto packThreads = [](int KT, int NT) { return KT * NT * 32; };
  // --- pack weights into fragment layout (f16) ---
  { int t = packThreads(39, 32);
    pack_b_kernel<<<(t + 255) / 256, 256, 0, stream>>>(W_in, (half_t*)(W + o_WinP), 39, 32, 1248, 512, 512, 0); }
  { int t = packThreads(16, 128);
    pack_b_kernel<<<(t + 255) / 256, 256, 0, stream>>>(eWih, (half_t*)(W + o_eWihP), 16, 128, 512, 2048, 512, 1);
    pack_b_kernel<<<(t + 255) / 256, 256, 0, stream>>>(eWhh, (half_t*)(W + o_eWhhP), 16, 128, 512, 2048, 512, 1);
    pack_b_kernel<<<(t + 255) / 256, 256, 0, stream>>>(dWih, (half_t*)(W + o_dWihP), 16, 128, 512, 2048, 512, 1);
    pack_b_kernel<<<(t + 255) / 256, 256, 0, stream>>>(dWhh, (half_t*)(W + o_dWhhP), 16, 128, 512, 2048, 512, 1); }
  { int t = packThreads(16, 64);
    pack_b_kernel<<<(t + 255) / 256, 256, 0, stream>>>(cbook, (half_t*)(W + o_cbP), 16, 64, 512, 1024, 512, 1); }
  for (int i = 0; i < 7; ++i) {
    int NT = vpad[i] / 16, t = packThreads(16, NT);
    pack_b_kernel<<<(t + 255) / 256, 256, 0, stream>>>((const float*)d_in[19 + 2 * i],
        (half_t*)(W + o_hP[i]), 16, NT, 512, vocab[i], vocab[i], 0);
    head_bias_kernel<<<1, 256, 0, stream>>>((const float*)d_in[20 + 2 * i],
        (float*)(W + o_hB[i]), vocab[i], vpad[i]);
  }
  vecadd_kernel<<<8, 256, 0, stream>>>(ebih, ebhh, (float*)(W + o_encB), 2048);
  vecadd_kernel<<<8, 256, 0, stream>>>(dbih, dbhh, (float*)(W + o_decB), 2048);
  cbnorm_kernel<<<4, 256, 0, stream>>>(cbook, (float*)(W + o_cbn));

  // --- embedding gather + input projection ---
  { size_t t = (size_t)MROWS * 1248;
    gather_kernel<<<(unsigned)((t + 255) / 256), 256, 0, stream>>>(tokens,
        embp[0], embp[1], embp[2], embp[3], embp[4], embp[5], embp[6], (half_t*)(W + o_X0)); }
  gemm_kernel<<<dim3(MROWS / 16, 8), 128, 0, stream>>>((const half_t*)(W + o_X0),
      (const half_t*)(W + o_WinP), b_in, nullptr, (half_t*)(W + o_x16), 512, 39);

  // --- encoder: xW precompute + persistent recurrent kernel ---
  gemm_kernel<<<dim3(MROWS / 16, 32), 128, 0, stream>>>((const half_t*)(W + o_x16),
      (const half_t*)(W + o_eWihP), (const float*)(W + o_encB), (float*)(W + o_xW), nullptr, 2048, 16);
  lstm_kernel<<<32, 128, 0, stream>>>((const float*)(W + o_xW), (const half_t*)(W + o_eWhhP),
      (float*)(W + o_enc), hbuf, syncp, syncp + 1);

  // --- VQ ---
  { size_t t = (size_t)MROWS * DM;
    cvt_f16_kernel<<<(unsigned)((t + 255) / 256), 256, 0, stream>>>((const float*)(W + o_enc),
        (half_t*)(W + o_em16), 1); }
  vq_kernel<<<MROWS / 16, 128, 0, stream>>>((const half_t*)(W + o_em16),
      (const half_t*)(W + o_cbP), (const float*)(W + o_cbn), (int*)(W + o_inds));
  { size_t t = (size_t)MROWS * DM;
    quant_loss_kernel<<<(unsigned)((t + 255) / 256), 256, 0, stream>>>((const float*)(W + o_enc),
        cbook, (const int*)(W + o_inds), (half_t*)(W + o_q16), accp); }

  // --- decoder ---
  gemm_kernel<<<dim3(MROWS / 16, 32), 128, 0, stream>>>((const half_t*)(W + o_q16),
      (const half_t*)(W + o_dWihP), (const float*)(W + o_decB), (float*)(W + o_xW), nullptr, 2048, 16);
  (void)hipMemsetAsync(W + o_hbuf, 0, (size_t)16 * DM * 2, stream);
  lstm_kernel<<<32, 128, 0, stream>>>((const float*)(W + o_xW), (const half_t*)(W + o_dWhhP),
      (float*)(W + o_dec), hbuf, syncp, syncp + 1);

  // --- heads + cross entropy ---
  { size_t t = (size_t)MROWS * DM;
    cvt_f16_kernel<<<(unsigned)((t + 255) / 256), 256, 0, stream>>>((const float*)(W + o_dec),
        (half_t*)(W + o_d16), 0); }
  for (int i = 0; i < 7; ++i) {
    gemm_kernel<<<dim3(MROWS / 16, (vpad[i] + 63) / 64), 128, 0, stream>>>(
        (const half_t*)(W + o_d16), (const half_t*)(W + o_hP[i]),
        (const float*)(W + o_hB[i]), (float*)(W + o_log), nullptr, vpad[i], 16);
    ce_kernel<<<MROWS / 8, 256, 0, stream>>>((const float*)(W + o_log), vpad[i], tokens, i, accp + 1);
  }

  // --- outputs ---
  { size_t t = (size_t)MROWS * DM;
    output_kernel<<<(unsigned)((t + 255) / 256), 256, 0, stream>>>((const float*)(W + o_dec),
        accp, accp + 1, (float*)d_out); }
}
